// LSTM_38422777430356
// MI455X (gfx1250) — matrix operand order repaired and checked by reading, not compiled, not run on hardware
//
#include <hip/hip_runtime.h>
#include <hip/hip_bf16.h>

// ---------------------------------------------------------------------------
// MI455X (gfx1250) LSTM: bf16 WMMA GEMMs + TDM async A-tile staging + fp32 cell.
// B=128, T=256, IN=512, H=1024, 4H=4096.
// Wave tile 32x64 (2 M-tiles x 4 N-tiles) so each B fragment feeds 2 WMMAs:
// 32 FLOP per L2 byte of B, matched to L2 bandwidth at WMMA peak.
// ---------------------------------------------------------------------------

typedef __bf16 bf16_t;
typedef __attribute__((ext_vector_type(16))) __bf16 v16bf;
typedef __attribute__((ext_vector_type(8)))  float  v8f;
typedef unsigned int __attribute__((ext_vector_type(4))) v4u;
typedef int          __attribute__((ext_vector_type(8))) v8i_t;
typedef int          __attribute__((ext_vector_type(4))) v4i_t;

#define BB   128
#define TT   256
#define INSZ 512
#define HH   1024
#define GG   4096   // 4*H
#define NC   10

// LDS A-tile: 32 rows x 64 bf16, TDM-padded to 144B row stride (72 elems)
#define AROW 72

union FragBF {
    v16bf v;
    uint4 u[2];
};

// ---------------------------------------------------------------------------
// TDM issue: load a 32-row x 64-col bf16 tile (row stride = stride_elems)
// from global into LDS at byte offset lds_off, padding +16B per 128B row
// so LDS rows land at 144B stride. D# per CDNA5 ISA ch.8.
// ---------------------------------------------------------------------------
__device__ __forceinline__ void tdm_load_a_tile(const bf16_t* gptr,
                                                int stride_elems,
                                                unsigned int lds_off) {
    unsigned long long ga = (unsigned long long)(uintptr_t)gptr;
    v4u g0;
    g0.x = 1u;                                   // count=1 (valid), user mode
    g0.y = lds_off;                              // lds_addr (bytes)
    g0.z = (unsigned int)(ga & 0xffffffffull);   // global_addr[31:0]
    g0.w = (unsigned int)((ga >> 32) & 0x1ffffffull) | (2u << 30); // [56:32], type=2

    v8i_t g1;
    // data_size=1 (2B), pad_enable=1, pad_interval=4 (32 DWORDs=128B),
    // pad_amount=3 (4 DWORDs=16B)
    g1[0] = (1 << 16) | (1 << 20) | (4 << 22) | (3 << 25);
    g1[1] = (int)((64u & 0xffffu) << 16);        // tensor_dim0[15:0] at [63:48]
    g1[2] = (int)(((64u >> 16) & 0xffffu) | ((32u & 0xffffu) << 16)); // dim0 hi | dim1 lo=32
    g1[3] = (int)(64u << 16);                    // tensor_dim1 hi=0 | tile_dim0=64
    g1[4] = (int)(32u);                          // tile_dim1=32, tile_dim2=0
    g1[5] = stride_elems;                        // tensor_dim0_stride[31:0]
    g1[6] = 0;                                   // stride hi | dim1_stride lo
    g1[7] = 0;

    v4i_t z4 = {0, 0, 0, 0};
#if __has_include(<hip/amd_detail/amd_gfx1250_TDM.h>)
    v8i_t z8 = {0, 0, 0, 0, 0, 0, 0, 0};
    __builtin_amdgcn_tensor_load_to_lds(g0, g1, z4, z4, z8, 0);
#else
    __builtin_amdgcn_tensor_load_to_lds(g0, g1, z4, z4, 0);
#endif
}

// ---------------------------------------------------------------------------
// Prep kernels
// ---------------------------------------------------------------------------
__global__ void cvt_f32_to_bf16(const float* __restrict__ src,
                                bf16_t* __restrict__ dst, long n) {
    long i = (long)blockIdx.x * blockDim.x + threadIdx.x;
    if (i < n) dst[i] = (bf16_t)src[i];
}

// Pack [w_ih | w_hh] row-major into bf16: out[g*K + k]. grid = (K/256, GG).
__global__ void pack_weights(const float* __restrict__ w_ih,
                             const float* __restrict__ w_hh,
                             bf16_t* __restrict__ out, int K0, int K) {
    int k = blockIdx.x * blockDim.x + threadIdx.x;
    int g = blockIdx.y;
    float v = (k < K0) ? w_ih[(long)g * K0 + k]
                       : w_hh[(long)g * HH + (k - K0)];
    out[(long)g * K + k] = (bf16_t)v;
}

__global__ void add_bias(const float* __restrict__ b_ih,
                         const float* __restrict__ b_hh,
                         float* __restrict__ out) {
    int i = blockIdx.x * blockDim.x + threadIdx.x;
    if (i < GG) out[i] = b_ih[i] + b_hh[i];
}

// ---------------------------------------------------------------------------
// gates = [in0_t | h_prev] @ W^T + bias   via v_wmma_f32_16x16x32_bf16.
//   M = 128 (batch), N = 4096 (gates), K = K0 + 1024
// Block = 128 threads (4 waves); block tile 32(M) x 256(N); wave = 32x64.
// A tile staged to LDS by TDM (double buffered, TENSORcnt-synchronized);
// B streams from L2 via global_load_b128 (+prefetch), reused across 2 M-tiles.
// Grid = (4096/256, 128/32) = (16, 4).
// ---------------------------------------------------------------------------
__global__ __launch_bounds__(128)
void lstm_gates_wmma(const bf16_t* __restrict__ in0, int in0_stride, int K0,
                     const bf16_t* __restrict__ hbf,   // [128 x 1024] bf16
                     const bf16_t* __restrict__ W,     // [4096 x K]   bf16 row-major
                     const float*  __restrict__ bias,  // [4096]
                     float*        __restrict__ gates, // [128 x 4096] f32
                     int K) {
    __shared__ __align__(16) bf16_t atile[2][32 * AROW];

    const int lane   = threadIdx.x & 31;
    const int wave   = threadIdx.x >> 5;
    const int laneHi = lane >> 4;     // 0 for lanes 0-15, 1 for 16-31
    const int lane16 = lane & 15;

    const int m0 = blockIdx.y * 32;               // batch tile (2 x 16)
    const int n0 = blockIdx.x * 256 + wave * 64;  // gate-col base (4 n-tiles)

    v8f acc[2][4];
#pragma unroll
    for (int tn = 0; tn < 4; ++tn) {
        float bv = bias[n0 + tn * 16 + lane16];
        acc[0][tn] = (v8f){bv, bv, bv, bv, bv, bv, bv, bv};
        acc[1][tn] = acc[0][tn];
    }

    // ISA 16-bit A 16x32 layout: lanes 0-15 hold k [0..8)+[16..24),
    // lanes 16-31 hold k [8..16)+[24..32). Row m = lane&15.
    const int kShiftA = laneHi * 8;
    // B 32x16 layout: lanes 0-15 hold k [0..8)+[16..24), lanes 16-31 hold k [8..16)+[24..32).
    const int kShiftB = laneHi * 8;

    const int nChunks = K >> 6;   // 64-wide k-chunks; K0 % 64 == 0

    // chunk ci source pointer/stride (uniform per chunk)
    auto chunk_src = [&](int ci, const bf16_t*& p, int& s) {
        int kc = ci << 6;
        if (kc < K0) { p = in0 + (long)m0 * in0_stride + kc; s = in0_stride; }
        else         { p = hbf + (long)m0 * HH + (kc - K0); s = HH; }
    };

    // prologue: DMA chunk 0 into buf 0
    if (wave == 0) {
        const bf16_t* p; int s;
        chunk_src(0, p, s);
        tdm_load_a_tile(p, s, (unsigned int)(uintptr_t)&atile[0][0]);
    }

    for (int ci = 0; ci < nChunks; ++ci) {
        const int pbuf = ci & 1;
        __syncthreads();   // all waves done reading buf[1-pbuf] (chunk ci-1) before refill
        if (wave == 0) {
            if (ci + 1 < nChunks) {
                const bf16_t* p; int s;
                chunk_src(ci + 1, p, s);
                tdm_load_a_tile(p, s,
                                (unsigned int)(uintptr_t)&atile[1 - pbuf][0]);
                __builtin_amdgcn_s_wait_tensorcnt(1);  // chunk ci complete
            } else {
                __builtin_amdgcn_s_wait_tensorcnt(0);
            }
        }
        __syncthreads();   // buf[pbuf] ready for all waves

#pragma unroll
        for (int ks = 0; ks < 64; ks += 32) {
            // two A fragments (M-tiles m0+0..15 and m0+16..31) from LDS
            FragBF a[2];
#pragma unroll
            for (int mt = 0; mt < 2; ++mt) {
                const bf16_t* alds =
                    &atile[pbuf][(mt * 16 + lane16) * AROW + ks + kShiftA];
                a[mt].u[0] = *(const uint4*)(alds);        // ds_load_b128
                a[mt].u[1] = *(const uint4*)(alds + 16);
            }
            const int kg = (ci << 6) + ks;
#pragma unroll
            for (int tn = 0; tn < 4; ++tn) {
                const bf16_t* brow =
                    W + (long)(n0 + tn * 16 + lane16) * K + kg + kShiftB;
                FragBF b;
                b.u[0] = ((const uint4*)brow)[0];
                b.u[1] = ((const uint4*)brow)[2];
                __builtin_prefetch(brow + 64, 0, 1);  // next k-chunk of B
                acc[0][tn] = __builtin_amdgcn_wmma_f32_16x16x32_bf16(
                    false, a[0].v, false, b.v, (short)0, acc[0][tn], false, false);
                acc[1][tn] = __builtin_amdgcn_wmma_f32_16x16x32_bf16(
                    false, a[1].v, false, b.v, (short)0, acc[1][tn], false, false);
            }
        }
    }

    // D layout: lane L, vgpr v -> m = v + 8*laneHi, n = L&15
#pragma unroll
    for (int mt = 0; mt < 2; ++mt) {
#pragma unroll
        for (int tn = 0; tn < 4; ++tn) {
            int n = n0 + tn * 16 + lane16;
#pragma unroll
            for (int v = 0; v < 8; ++v) {
                int m = m0 + mt * 16 + v + laneHi * 8;
                gates[(long)m * GG + n] = acc[mt][tn][v];
            }
        }
    }
}

// ---------------------------------------------------------------------------
// LSTM cell pointwise: c,h update (fp32 math, bf16 h for next GEMM)
// ---------------------------------------------------------------------------
__global__ __launch_bounds__(256)
void lstm_cell(const float* __restrict__ gates,   // [128 x 4096]
               float*       __restrict__ c,       // [128 x 1024]
               bf16_t*      __restrict__ hbf,     // [128 x 1024]
               float*       __restrict__ hf32) {  // optional [128 x 1024]
    int idx = blockIdx.x * blockDim.x + threadIdx.x;  // b*H + j
    int b = idx >> 10;
    int j = idx & (HH - 1);
    const float* g = gates + (long)b * GG;
    float gi = g[j];
    float gf = g[HH + j];
    float gg = g[2 * HH + j];
    float go = g[3 * HH + j];
    float si = 1.f / (1.f + __expf(-gi));
    float sf = 1.f / (1.f + __expf(-gf));
    float so = 1.f / (1.f + __expf(-go));
    float cn = sf * c[idx] + si * tanhf(gg);
    float h  = so * tanhf(cn);
    c[idx]   = cn;
    hbf[idx] = (bf16_t)h;
    if (hf32) hf32[idx] = h;
}

// ---------------------------------------------------------------------------
// Head: out = relu(last @ w1^T + b1) @ w2^T + b2   (128x1024 -> 32 -> 10)
// ---------------------------------------------------------------------------
__global__ __launch_bounds__(64)
void head_kernel(const float* __restrict__ last,   // [128 x 1024]
                 const float* __restrict__ w1, const float* __restrict__ b1,
                 const float* __restrict__ w2, const float* __restrict__ b2,
                 float* __restrict__ out) {        // [128 x 10]
    __shared__ float hid[32];
    int b   = blockIdx.x;
    int tid = threadIdx.x;
    if (tid < 32) {
        const float* row = last + (long)b * HH;
        const float* w   = w1 + (long)tid * HH;
        float s = b1[tid];
        for (int k = 0; k < HH; ++k) s += row[k] * w[k];
        hid[tid] = fmaxf(s, 0.f);
    }
    __syncthreads();
    if (tid < NC) {
        float s = b2[tid];
        const float* w = w2 + tid * 32;
        for (int k = 0; k < 32; ++k) s += hid[k] * w[k];
        out[(long)b * NC + tid] = s;
    }
}

// ---------------------------------------------------------------------------
// Host launch
// ---------------------------------------------------------------------------
static inline size_t align256(size_t x) { return (x + 255) & ~(size_t)255; }

extern "C" void kernel_launch(void* const* d_in, const int* in_sizes, int n_in,
                              void* d_out, int out_size, void* d_ws, size_t ws_size,
                              hipStream_t stream) {
    (void)in_sizes; (void)n_in; (void)out_size; (void)ws_size;

    const float* x = (const float*)d_in[0];
    const float* w_ih[3] = {(const float*)d_in[1], (const float*)d_in[5], (const float*)d_in[9]};
    const float* w_hh[3] = {(const float*)d_in[2], (const float*)d_in[6], (const float*)d_in[10]};
    const float* b_ih[3] = {(const float*)d_in[3], (const float*)d_in[7], (const float*)d_in[11]};
    const float* b_hh[3] = {(const float*)d_in[4], (const float*)d_in[8], (const float*)d_in[12]};
    const float* w1 = (const float*)d_in[13];
    const float* b1 = (const float*)d_in[14];
    const float* w2 = (const float*)d_in[15];
    const float* b2 = (const float*)d_in[16];
    float* out = (float*)d_out;

    const int Kl[3]  = {INSZ + HH, HH + HH, HH + HH};  // 1536, 2048, 2048
    const int K0l[3] = {INSZ, HH, HH};

    // Workspace carve-up
    char* ws = (char*)d_ws;
    size_t off = 0;
    bf16_t* xbf = (bf16_t*)(ws + off); off = align256(off + (size_t)BB * TT * INSZ * 2);
    bf16_t* Wl[3];
    for (int l = 0; l < 3; ++l) { Wl[l] = (bf16_t*)(ws + off); off = align256(off + (size_t)GG * Kl[l] * 2); }
    float* biasl[3];
    for (int l = 0; l < 3; ++l) { biasl[l] = (float*)(ws + off); off = align256(off + (size_t)GG * 4); }
    float* cst = (float*)(ws + off);   // 3 x [128x1024] f32 (zeroed each call)
    off = align256(off + (size_t)3 * BB * HH * 4);
    bf16_t* hbf = (bf16_t*)(ws + off); // 3 x [128x1024] bf16 (zeroed each call)
    off = align256(off + (size_t)3 * BB * HH * 2);
    float* gates = (float*)(ws + off); off = align256(off + (size_t)BB * GG * 4);
    float* h2f   = (float*)(ws + off); off = align256(off + (size_t)BB * HH * 4);

    // --- prep ---
    {
        long n = (long)BB * TT * INSZ;
        cvt_f32_to_bf16<<<(unsigned)((n + 255) / 256), 256, 0, stream>>>(x, xbf, n);
    }
    for (int l = 0; l < 3; ++l) {
        dim3 pg(Kl[l] / 256, GG);
        pack_weights<<<pg, 256, 0, stream>>>(w_ih[l], w_hh[l], Wl[l], K0l[l], Kl[l]);
        add_bias<<<GG / 256, 256, 0, stream>>>(b_ih[l], b_hh[l], biasl[l]);
    }
    // zero h, c initial state (ws is poisoned; must re-zero every call)
    hipMemsetAsync(cst, 0, (size_t)3 * BB * HH * 4, stream);
    hipMemsetAsync(hbf, 0, (size_t)3 * BB * HH * 2, stream);

    // --- recurrence ---
    dim3 ggrid(GG / 256, BB / 32);  // (16, 4)
    for (int t = 0; t < TT; ++t) {
        for (int l = 0; l < 3; ++l) {
            const bf16_t* in0;
            int stride;
            if (l == 0) { in0 = xbf + (long)t * INSZ; stride = TT * INSZ; }
            else        { in0 = hbf + (long)(l - 1) * BB * HH; stride = HH; }
            bf16_t* hb = hbf + (long)l * BB * HH;
            float*  cc = cst + (long)l * BB * HH;

            lstm_gates_wmma<<<ggrid, 128, 0, stream>>>(
                in0, stride, K0l[l], hb, Wl[l], biasl[l], gates, Kl[l]);

            float* hw = (l == 2 && t == TT - 1) ? h2f : nullptr;
            lstm_cell<<<(BB * HH) / 256, 256, 0, stream>>>(gates, cc, hb, hw);
        }
    }

    // --- head ---
    head_kernel<<<BB, 64, 0, stream>>>(h2f, w1, b1, w2, b2, out);
}
